// MultimodalMambaFusion_15831249453205
// MI455X (gfx1250) — matrix operand order repaired and checked by reading, not compiled, not run on hardware
//
#include <hip/hip_runtime.h>
#include <hip/hip_bf16.h>

#define DMODEL   576
#define DEPTH    8
#define DINNER   1152
#define DSTATE   16
#define DCONV    4
#define DTRANK   36
#define NCLS     1000
#define BSZ      2
#define SEQL     1185            // 1 + 784 + 400
#define MROWS    (BSZ * SEQL)    // 2370
#define MTILES   149             // ceil(2370/16)
#define MP       (MTILES * 16)   // 2384 padded rows
#define DBLPAD   80              // 68 padded to 5x16
#define XPROJN   68              // DTRANK + 2*DSTATE
#define SCAND    288             // channels per scan block (9 waves)

typedef __bf16 bf16_t;
typedef bf16_t v16bf __attribute__((ext_vector_type(16)));
typedef bf16_t v8bf  __attribute__((ext_vector_type(8)));
typedef float  v8f   __attribute__((ext_vector_type(8)));

union Frag16 { v16bf v; v8bf h[2]; };
union FragF8 { v8f v; float e[8]; };

__device__ __forceinline__ float dev_silu(float x) { return x / (1.f + __expf(-x)); }

// ---------------------------------------------------------------------------
// bf16-WMMA GEMM: C[M,N] = A[M,K] * W[N,K]^T, A/W pre-converted to bf16.
// One wave computes a 16 x (NTW*16) strip.  Software-pipelined: fragments for
// k-step k0+32 are loaded before the WMMAs consuming k-step k0, so loads stay
// in flight behind the matrix ops (staged s_wait_loadcnt, no full drains).
//  A frag (16x32): lane m = lane&15, khalf = lane>>4;
//      runs K = k0+khalf*8+[0..7]  and  K = k0+16+khalf*8+[0..7]
//  B frag (32x16): lane n = lane&15, khalf = lane>>4;
//      runs K = k0+khalf*8+[0..7]  and  K = k0+16+khalf*8+[0..7]
// ---------------------------------------------------------------------------
template <int NTW>
__global__ void wmma_gemm_bf16(const bf16_t* __restrict__ A, int lda,
                               const bf16_t* __restrict__ W, int ldw,
                               float* __restrict__ C, int ldc,
                               int Mreal, int K, int nWaveCols, int totalWaves) {
    int wave = (int)((blockIdx.x * blockDim.x + threadIdx.x) >> 5);
    int lane = (int)(threadIdx.x & 31);
    if (wave >= totalWaves) return;
    int tm = wave / nWaveCols;
    int tc = wave - tm * nWaveCols;
    int n0 = tc * (NTW * 16);

    int mA = tm * 16 + (lane & 15);
    if (mA >= Mreal) mA = Mreal - 1;                 // clamp: masked at store
    const bf16_t* Arow = A + (size_t)mA * lda + (lane >> 4) * 8;

    const bf16_t* Brow[NTW];
    #pragma unroll
    for (int s = 0; s < NTW; ++s) {
        int n = n0 + s * 16 + (lane & 15);
        Brow[s] = W + (size_t)n * ldw + (lane >> 4) * 8;
    }

    v8f acc[NTW];
    #pragma unroll
    for (int s = 0; s < NTW; ++s) acc[s] = (v8f){};

    // prologue: fragments for k0 = 0
    Frag16 aCur, bCur[NTW];
    aCur.h[0] = *(const v8bf*)(Arow);
    aCur.h[1] = *(const v8bf*)(Arow + 16);
    #pragma unroll
    for (int s = 0; s < NTW; ++s) {
        bCur[s].h[0] = *(const v8bf*)(Brow[s]);
        bCur[s].h[1] = *(const v8bf*)(Brow[s] + 16);
    }

    for (int k0 = 0; k0 < K; k0 += 32) {
        int k1 = (k0 + 32 < K) ? (k0 + 32) : 0;      // wrap: avoids OOB tail
        Frag16 aNxt, bNxt[NTW];
        aNxt.h[0] = *(const v8bf*)(Arow + k1);
        aNxt.h[1] = *(const v8bf*)(Arow + k1 + 16);
        #pragma unroll
        for (int s = 0; s < NTW; ++s) {
            bNxt[s].h[0] = *(const v8bf*)(Brow[s] + k1);
            bNxt[s].h[1] = *(const v8bf*)(Brow[s] + k1 + 16);
        }
        #pragma unroll
        for (int s = 0; s < NTW; ++s)
            acc[s] = __builtin_amdgcn_wmma_f32_16x16x32_bf16(false, aCur.v, false,
                                                             bCur[s].v, (short)0,
                                                             acc[s], false, false);
        aCur = aNxt;
        #pragma unroll
        for (int s = 0; s < NTW; ++s) bCur[s] = bNxt[s];
    }

    // C/D layout: lane n = lane&15, rows mBase..mBase+7 across the 8 VGPRs
    int mBase = tm * 16 + (lane >> 4) * 8;
    #pragma unroll
    for (int s = 0; s < NTW; ++s) {
        FragF8 r; r.v = acc[s];
        int nn = n0 + s * 16 + (lane & 15);
        #pragma unroll
        for (int j = 0; j < 8; ++j) {
            int mm = mBase + j;
            if (mm < Mreal) C[(size_t)mm * ldc + nn] = r.e[j];
        }
    }
}

// ---------------------------------------------------------------------------
// Elementwise f32 -> bf16 weight mirror.
// ---------------------------------------------------------------------------
__global__ void cvt_bf16(const float* __restrict__ in, bf16_t* __restrict__ out, int n) {
    int i = (int)(blockIdx.x * blockDim.x + threadIdx.x);
    if (i < n) out[i] = (bf16_t)in[i];
}

// ---------------------------------------------------------------------------
// x_proj weights: (8,68,1152) f32 -> (8,80,1152) bf16 with zero pad rows.
// ---------------------------------------------------------------------------
__global__ void pad_xproj_bf16(const float* __restrict__ in, bf16_t* __restrict__ out) {
    int i = (int)(blockIdx.x * blockDim.x + threadIdx.x);
    if (i >= DEPTH * DBLPAD * DINNER) return;
    int p = i / (DBLPAD * DINNER);
    int rem = i - p * (DBLPAD * DINNER);
    int n = rem / DINNER, k = rem - n * DINNER;
    out[i] = (n < XPROJN) ? (bf16_t)in[((size_t)p * XPROJN + n) * DINNER + k] : (bf16_t)0.f;
}

// ---------------------------------------------------------------------------
// Build fused sequence: cls mix + modality embeddings.  Also zero residual.
// ---------------------------------------------------------------------------
__global__ void build_x(const float* __restrict__ vf, const float* __restrict__ sf,
                        const float* __restrict__ vme, const float* __restrict__ sme,
                        const float* __restrict__ alpha_p,
                        float* __restrict__ h, float* __restrict__ r) {
    int idx = (int)(blockIdx.x * blockDim.x + threadIdx.x);
    if (idx >= MROWS * DMODEL) return;
    int row = idx / DMODEL, c = idx - row * DMODEL;
    int b = row / SEQL, l = row - b * SEQL;
    float val;
    if (l == 0) {
        float a = 1.f / (1.f + __expf(-alpha_p[0]));
        val = vf[((size_t)b * 785) * DMODEL + c] * a +
              sf[((size_t)b * 401) * DMODEL + c] * (1.f - a);
    } else if (l <= 784) {
        val = vf[((size_t)b * 785 + l) * DMODEL + c] + vme[c];
    } else {
        val = sf[((size_t)b * 401 + (l - 784)) * DMODEL + c] + sme[c];
    }
    h[idx] = val;
    r[idx] = 0.f;
}

// ---------------------------------------------------------------------------
// r += h ; hn = (bf16)(rmsnorm(r) * w).   One wave32 per row (576 = 18*32).
// ---------------------------------------------------------------------------
__global__ void residual_rmsnorm(float* __restrict__ rbuf, const float* __restrict__ hbuf,
                                 const float* __restrict__ w, bf16_t* __restrict__ hn,
                                 int Mreal) {
    int wid  = (int)((blockIdx.x * blockDim.x + threadIdx.x) >> 5);
    int lane = (int)(threadIdx.x & 31);
    if (wid >= Mreal) return;
    size_t base = (size_t)wid * DMODEL;
    float v[18];
    float ss = 0.f;
    #pragma unroll
    for (int i = 0; i < 18; ++i) {
        float x = rbuf[base + lane + i * 32] + hbuf[base + lane + i * 32];
        v[i] = x;
        ss += x * x;
    }
    #pragma unroll
    for (int m = 16; m >= 1; m >>= 1) ss += __shfl_xor(ss, m, 32);
    float scale = rsqrtf(ss / (float)DMODEL + 1e-5f);
    #pragma unroll
    for (int i = 0; i < 18; ++i) {
        int c = lane + i * 32;
        rbuf[base + c] = v[i];
        hn[base + c]   = (bf16_t)(v[i] * scale * w[c]);
    }
}

// ---------------------------------------------------------------------------
// Causal depthwise conv (K=4) + SiLU.  dirBwd folds the sequence reversal so
// outputs stay in original l indexing.  Writes f32 (scan) + bf16 (GEMM A).
// ---------------------------------------------------------------------------
__global__ void conv_silu(const float* __restrict__ xz, const float* __restrict__ cw,
                          const float* __restrict__ cb, float* __restrict__ xc,
                          bf16_t* __restrict__ xcbf, int dirBwd) {
    int idx = (int)(blockIdx.x * blockDim.x + threadIdx.x);
    if (idx >= MROWS * DINNER) return;
    int row = idx / DINNER, d = idx - row * DINNER;
    int b = row / SEQL, l = row - b * SEQL;
    float acc = cb[d];
    #pragma unroll
    for (int k = 0; k < DCONV; ++k) {
        int lx = dirBwd ? (l + 3 - k) : (l + k - 3);
        if (lx >= 0 && lx < SEQL)
            acc += cw[d * DCONV + k] * xz[((size_t)(b * SEQL + lx)) * (2 * DINNER) + d];
    }
    float s = dev_silu(acc);
    xc[(size_t)row * DINNER + d]   = s;
    xcbf[(size_t)row * DINNER + d] = (bf16_t)s;
}

// ---------------------------------------------------------------------------
// Selective scan: one thread per (b, d) channel, 16-wide state in registers,
// per-step dt/B/C row staged through LDS, dt_w tile cached in LDS.
// Fuses delta GEMM (K=36) + softplus + scan + D-term + SiLU(z) gating.
// Output accumulated in bf16 (feeds out_proj WMMA directly).
// ---------------------------------------------------------------------------
__global__ void scan_kernel(const float* __restrict__ xc, const float* __restrict__ dbl,
                            const float* __restrict__ xz,
                            const float* __restrict__ dtw, const float* __restrict__ dtb,
                            const float* __restrict__ alog, const float* __restrict__ dssm,
                            bf16_t* __restrict__ ycomb, int dirBwd) {
    __shared__ float s_dtw[SCAND * DTRANK];   // 41472 B
    __shared__ float s_row[DBLPAD];           // dt(36) + B(16) + C(16)
    int dBase = (int)blockIdx.x * SCAND;
    int b = (int)blockIdx.y;
    int t = (int)threadIdx.x;
    int d = dBase + t;

    for (int i = t; i < SCAND * DTRANK; i += SCAND)
        s_dtw[i] = dtw[(size_t)dBase * DTRANK + i];

    float A[DSTATE], h[DSTATE];
    #pragma unroll
    for (int n = 0; n < DSTATE; ++n) {
        A[n] = -__expf(alog[(size_t)d * DSTATE + n]);
        h[n] = 0.f;
    }
    float bias = dtb[d];
    float Dp   = dssm[d];
    const float* wrow = &s_dtw[t * DTRANK];
    __syncthreads();

    for (int step = 0; step < SEQL; ++step) {
        int l = dirBwd ? (SEQL - 1 - step) : step;
        size_t row = (size_t)b * SEQL + l;
        __syncthreads();                       // previous-step readers done
        if (t < DTRANK + 2 * DSTATE) s_row[t] = dbl[row * DBLPAD + t];
        __syncthreads();

        float x = bias;
        #pragma unroll 6
        for (int r2 = 0; r2 < DTRANK; ++r2) x += s_row[r2] * wrow[r2];
        float delta = (x > 20.f) ? x : __logf(1.f + __expf(x));

        float u  = xc[row * DINNER + d];
        float du = delta * u;
        float y  = 0.f;
        #pragma unroll
        for (int n = 0; n < DSTATE; ++n) {
            h[n] = __expf(delta * A[n]) * h[n] + du * s_row[DTRANK + n];
            y   += h[n] * s_row[DTRANK + DSTATE + n];
        }
        float z   = xz[row * (2 * DINNER) + DINNER + d];
        float out = (y + u * Dp) * dev_silu(z);
        size_t oi = row * DINNER + d;
        if (dirBwd) out += (float)ycomb[oi];
        ycomb[oi] = (bf16_t)out;
    }
}

// ---------------------------------------------------------------------------
// Classification head: logits[b,c] = hn[b, l=0, :] . head_w[c,:] + head_b[c]
// ---------------------------------------------------------------------------
__global__ void head_kernel(const bf16_t* __restrict__ hn, const float* __restrict__ hw,
                            const float* __restrict__ hb, float* __restrict__ out) {
    int idx = (int)(blockIdx.x * blockDim.x + threadIdx.x);
    if (idx >= BSZ * NCLS) return;
    int b = idx / NCLS, c = idx - b * NCLS;
    const bf16_t* x = hn + (size_t)b * SEQL * DMODEL;
    const float*  w = hw + (size_t)c * DMODEL;
    float acc = hb[c];
    for (int k = 0; k < DMODEL; ++k) acc += (float)x[k] * w[k];
    out[idx] = acc;
}

static inline int cdiv(int a, int b) { return (a + b - 1) / b; }
static inline size_t alignup(size_t x) { return (x + 255) & ~(size_t)255; }

extern "C" void kernel_launch(void* const* d_in, const int* in_sizes, int n_in,
                              void* d_out, int out_size, void* d_ws, size_t ws_size,
                              hipStream_t stream) {
    (void)in_sizes; (void)n_in; (void)out_size; (void)ws_size;
    const float* vf       = (const float*)d_in[0];
    const float* sf       = (const float*)d_in[1];
    const float* vme      = (const float*)d_in[2];
    const float* sme      = (const float*)d_in[3];
    const float* alpha    = (const float*)d_in[4];
    const float* in_proj  = (const float*)d_in[5];
    const float* conv_w   = (const float*)d_in[6];
    const float* conv_b   = (const float*)d_in[7];
    const float* xproj_w  = (const float*)d_in[8];
    const float* dt_w     = (const float*)d_in[9];
    const float* dt_b     = (const float*)d_in[10];
    const float* A_log    = (const float*)d_in[11];
    const float* D_ssm    = (const float*)d_in[12];
    const float* conv_wb  = (const float*)d_in[13];
    const float* conv_bb  = (const float*)d_in[14];
    const float* xproj_wb = (const float*)d_in[15];
    const float* dt_wb    = (const float*)d_in[16];
    const float* dt_bb    = (const float*)d_in[17];
    const float* A_logb   = (const float*)d_in[18];
    const float* D_ssmb   = (const float*)d_in[19];
    const float* out_proj = (const float*)d_in[20];
    const float* norm_w   = (const float*)d_in[21];
    const float* norm_f_w = (const float*)d_in[22];
    const float* head_w   = (const float*)d_in[23];
    const float* head_b   = (const float*)d_in[24];
    float* out = (float*)d_out;

    // -------- workspace carving (256B aligned) --------
    char* wp = (char*)d_ws;
    auto carve = [&](size_t bytes) { char* p = wp; wp += alignup(bytes); return p; };
    float*  h_buf   = (float*) carve(sizeof(float)  * MP * DMODEL);
    float*  r_buf   = (float*) carve(sizeof(float)  * MP * DMODEL);
    float*  xz_buf  = (float*) carve(sizeof(float)  * MP * 2 * DINNER);
    float*  xcf     = (float*) carve(sizeof(float)  * MP * DINNER);
    float*  xcb     = (float*) carve(sizeof(float)  * MP * DINNER);
    float*  dblf    = (float*) carve(sizeof(float)  * MP * DBLPAD);
    float*  dblb    = (float*) carve(sizeof(float)  * MP * DBLPAD);
    bf16_t* hn_bf   = (bf16_t*)carve(sizeof(bf16_t) * MP * DMODEL);
    bf16_t* xcf_bf  = (bf16_t*)carve(sizeof(bf16_t) * MP * DINNER);
    bf16_t* xcb_bf  = (bf16_t*)carve(sizeof(bf16_t) * MP * DINNER);
    bf16_t* ycb_bf  = (bf16_t*)carve(sizeof(bf16_t) * MP * DINNER);
    bf16_t* w_in_bf = (bf16_t*)carve(sizeof(bf16_t) * DEPTH * 2 * DINNER * DMODEL);
    bf16_t* w_out_bf= (bf16_t*)carve(sizeof(bf16_t) * DEPTH * DMODEL * DINNER);
    bf16_t* w_xpf_bf= (bf16_t*)carve(sizeof(bf16_t) * DEPTH * DBLPAD * DINNER);
    bf16_t* w_xpb_bf= (bf16_t*)carve(sizeof(bf16_t) * DEPTH * DBLPAD * DINNER);

    // -------- one-time (per launch) weight mirrors --------
    {
        int nIn = DEPTH * 2 * DINNER * DMODEL;
        cvt_bf16<<<cdiv(nIn, 256), 256, 0, stream>>>(in_proj, w_in_bf, nIn);
        int nOut = DEPTH * DMODEL * DINNER;
        cvt_bf16<<<cdiv(nOut, 256), 256, 0, stream>>>(out_proj, w_out_bf, nOut);
        int nXp = DEPTH * DBLPAD * DINNER;
        pad_xproj_bf16<<<cdiv(nXp, 256), 256, 0, stream>>>(xproj_w,  w_xpf_bf);
        pad_xproj_bf16<<<cdiv(nXp, 256), 256, 0, stream>>>(xproj_wb, w_xpb_bf);
    }

    // 0) build fused sequence, zero residual
    build_x<<<cdiv(MROWS * DMODEL, 256), 256, 0, stream>>>(vf, sf, vme, sme, alpha,
                                                           h_buf, r_buf);

    for (int p = 0; p < DEPTH; ++p) {
        const bf16_t* W_in  = w_in_bf  + (size_t)p * 2 * DINNER * DMODEL;
        const bf16_t* W_out = w_out_bf + (size_t)p * DMODEL * DINNER;
        const float*  nw    = norm_w   + (size_t)p * DMODEL;

        // 1) residual add + rmsnorm (bf16 out)
        residual_rmsnorm<<<cdiv(MROWS, 8), 256, 0, stream>>>(r_buf, h_buf, nw, hn_bf, MROWS);

        // 2) in_proj: [M,576] x [2304,576]^T -> xz[M,2304]   (144 N-tiles, 4/wave)
        {
            int nWC = (2 * DINNER / 16) / 4, tot = MTILES * nWC;
            wmma_gemm_bf16<4><<<cdiv(tot, 8), 256, 0, stream>>>(
                hn_bf, DMODEL, W_in, DMODEL, xz_buf, 2 * DINNER,
                MROWS, DMODEL, nWC, tot);
        }

        // 3) causal dwconv + SiLU (fwd & folded-bwd), f32 + bf16 outputs
        conv_silu<<<cdiv(MROWS * DINNER, 256), 256, 0, stream>>>(
            xz_buf, conv_w  + (size_t)p * DINNER * DCONV, conv_b  + (size_t)p * DINNER,
            xcf, xcf_bf, 0);
        conv_silu<<<cdiv(MROWS * DINNER, 256), 256, 0, stream>>>(
            xz_buf, conv_wb + (size_t)p * DINNER * DCONV, conv_bb + (size_t)p * DINNER,
            xcb, xcb_bf, 1);

        // 4) x_proj: [M,1152] x [80(z68),1152]^T -> dbl[M,80]  (5 N-tiles / wave)
        {
            int tot = MTILES;
            wmma_gemm_bf16<5><<<cdiv(tot, 8), 256, 0, stream>>>(
                xcf_bf, DINNER, w_xpf_bf + (size_t)p * DBLPAD * DINNER, DINNER,
                dblf, DBLPAD, MROWS, DINNER, 1, tot);
            wmma_gemm_bf16<5><<<cdiv(tot, 8), 256, 0, stream>>>(
                xcb_bf, DINNER, w_xpb_bf + (size_t)p * DBLPAD * DINNER, DINNER,
                dblb, DBLPAD, MROWS, DINNER, 1, tot);
        }

        // 5) selective scans (fwd writes ycomb, bwd accumulates)
        {
            dim3 g(DINNER / SCAND, BSZ);
            scan_kernel<<<g, SCAND, 0, stream>>>(
                xcf, dblf, xz_buf,
                dt_w  + (size_t)p * DINNER * DTRANK, dt_b  + (size_t)p * DINNER,
                A_log + (size_t)p * DINNER * DSTATE, D_ssm + (size_t)p * DINNER, ycb_bf, 0);
            scan_kernel<<<g, SCAND, 0, stream>>>(
                xcb, dblb, xz_buf,
                dt_wb  + (size_t)p * DINNER * DTRANK, dt_bb  + (size_t)p * DINNER,
                A_logb + (size_t)p * DINNER * DSTATE, D_ssmb + (size_t)p * DINNER, ycb_bf, 1);
        }

        // 6) out_proj: [M,1152] x [576,1152]^T -> h[M,576]  (36 N-tiles, 4/wave)
        {
            int nWC = (DMODEL / 16) / 4, tot = MTILES * nWC;
            wmma_gemm_bf16<4><<<cdiv(tot, 8), 256, 0, stream>>>(
                ycb_bf, DINNER, W_out, DINNER, h_buf, DMODEL,
                MROWS, DINNER, nWC, tot);
        }
    }

    // final norm + head (row 0 of each batch)
    residual_rmsnorm<<<cdiv(MROWS, 8), 256, 0, stream>>>(r_buf, h_buf, norm_f_w, hn_bf, MROWS);
    head_kernel<<<cdiv(BSZ * NCLS, 256), 256, 0, stream>>>(hn_bf, head_w, head_b, out);
}
